// E74FullMatrixCell_23751169147415
// MI455X (gfx1250) — matrix operand order repaired and checked by reading, not compiled, not run on hardware
//
#include <hip/hip_runtime.h>
#include <hip/hip_bf16.h>

// Problem constants (fixed by the reference setup_inputs()).
#define TT  512
#define BB  16
#define DD  1024
#define NN  512
#define TB  (TT*BB)          // 8192 rows of x
#define TBN (TT*(long)BB*NN) // 4194304 elements per projection

typedef __attribute__((ext_vector_type(16))) __bf16 v16bf;
typedef __attribute__((ext_vector_type(8)))  float  v8f;

// -------- helpers --------------------------------------------------------

__device__ __forceinline__ float wave_reduce_sum(float x) {
    // wave32 butterfly reduction; all lanes end with the sum
    x += __shfl_xor(x, 16, 32);
    x += __shfl_xor(x,  8, 32);
    x += __shfl_xor(x,  4, 32);
    x += __shfl_xor(x,  2, 32);
    x += __shfl_xor(x,  1, 32);
    return x;
}

__device__ __forceinline__ float fast_tanh(float x) {
#if __has_builtin(__builtin_amdgcn_tanhf)
    // compiler-known trans op: hazard NOPs / s_delay_alu handled by backend
    return __builtin_amdgcn_tanhf(x);
#else
    // native gfx1250 tanh via asm; trailing v_nop satisfies the TRANS-op
    // "1 independent op before result use" rule (ISA 7.4) for the last use
    float y;
    asm volatile("v_tanh_f32 %0, %1\n\tv_nop" : "=v"(y) : "v"(x));
    return y;
#endif
}

struct bf16pair { __bf16 hi; __bf16 lo; };

__device__ __forceinline__ bf16pair split_bf16(float x) {
    bf16pair p;
    p.hi = (__bf16)x;
    p.lo = (__bf16)(x - (float)p.hi);
    return p;
}

// -------- phase 1: projections via bf16 split WMMA -----------------------
// C[m,n] = sum_d X[m,d] * W[n,d]   (M=8192, K=1024, N=512)
// One wave computes one 16x16 tile; 8 waves/block cover 16(M) x 128(N).
// grid = (M/16, N/128, 3) ; z selects {W_q->q, W_v->v, W_k->k}
__global__ __launch_bounds__(256)
void proj_kernel(const float* __restrict__ X,
                 const float* __restrict__ Wq,
                 const float* __restrict__ Wv,
                 const float* __restrict__ Wk,
                 float* __restrict__ qb,
                 float* __restrict__ vb,
                 float* __restrict__ kb) {
    const int lane = threadIdx.x & 31;
    const int wave = threadIdx.x >> 5;
    const int half = lane >> 4;          // 0 or 1
    const int l16  = lane & 15;

    const float* W; float* Dst;
    if (blockIdx.z == 0)      { W = Wq; Dst = qb; }
    else if (blockIdx.z == 1) { W = Wv; Dst = vb; }
    else                      { W = Wk; Dst = kb; }

    const int tileM = blockIdx.x * 16;
    const int tileN = blockIdx.y * 128 + wave * 16;

    const int m = tileM + l16;           // A row handled by this lane
    const int n = tileN + l16;           // B/C column handled by this lane
    const float* xrow = X + (long)m * DD;
    const float* wrow = W + (long)n * DD;

    v8f acc = {};

    for (int kc = 0; kc < DD; kc += 32) {
        v16bf a_hi, a_lo, b_hi, b_lo;
        // A (16x32 bf16): VGPR r holds K pair base = (r&4)*4 + (r&3)*2 + half*8
        #pragma unroll
        for (int r = 0; r < 8; ++r) {
            const int kk = kc + ((r & 4) << 2) + ((r & 3) << 1) + (half << 3);
            const float2 xv = *(const float2*)(xrow + kk);
            const bf16pair px = split_bf16(xv.x);
            const bf16pair py = split_bf16(xv.y);
            a_hi[2*r]   = px.hi;  a_lo[2*r]   = px.lo;
            a_hi[2*r+1] = py.hi;  a_lo[2*r+1] = py.lo;
        }
        // B (32x16 bf16): same K order as A: VGPR r holds K pair base = (r&4)*4 + (r&3)*2 + half*8
        #pragma unroll
        for (int r = 0; r < 8; ++r) {
            const int kk = kc + ((r & 4) << 2) + ((r & 3) << 1) + (half << 3);
            const float2 wv = *(const float2*)(wrow + kk);
            const bf16pair px = split_bf16(wv.x);
            const bf16pair py = split_bf16(wv.y);
            b_hi[2*r]   = px.hi;  b_lo[2*r]   = px.lo;
            b_hi[2*r+1] = py.hi;  b_lo[2*r+1] = py.lo;
        }
        // fp32-accurate product via hi*hi + hi*lo + lo*hi
        acc = __builtin_amdgcn_wmma_f32_16x16x32_bf16(false, a_hi, false, b_hi,
                                                      (short)0, acc, false, false);
        acc = __builtin_amdgcn_wmma_f32_16x16x32_bf16(false, a_hi, false, b_lo,
                                                      (short)0, acc, false, false);
        acc = __builtin_amdgcn_wmma_f32_16x16x32_bf16(false, a_lo, false, b_hi,
                                                      (short)0, acc, false, false);
    }

    // C/D layout: VGPR r -> row M = r + 8*half, col N = lane&15
    #pragma unroll
    for (int r = 0; r < 8; ++r) {
        const int mm = tileM + r + 8 * half;
        Dst[(long)mm * NN + n] = acc[r];
    }
}

// -------- phase 1b: normalize k rows (k / (||k|| + eps)) ------------------
// one wave per (t,b) row of length N=512 ; grid = TB/8 blocks of 256
__global__ __launch_bounds__(256)
void knorm_kernel(float* __restrict__ kb) {
    const int lane = threadIdx.x & 31;
    const int wave = threadIdx.x >> 5;
    const long row = (long)blockIdx.x * 8 + wave;
    float* p = kb + row * NN;

    float ss = 0.f;
    #pragma unroll
    for (int c = 0; c < 16; ++c) {
        const float x = p[c * 32 + lane];
        ss = fmaf(x, x, ss);
    }
    ss = wave_reduce_sum(ss);
    const float inv = 1.0f / (sqrtf(ss) + 1e-6f);
    #pragma unroll
    for (int c = 0; c < 16; ++c)
        p[c * 32 + lane] *= inv;
}

// -------- phase 2: sequential delta-rule scan ----------------------------
// One wave owns one (batch, row i) of S for all T steps; S row lives in
// 16 VGPRs per lane. Column mapping: j = c4*128 + lane*4 + e  (c = c4*4+e)
// grid = (N/8, B), block = 256 (8 waves)
__global__ __launch_bounds__(256)
void scan_kernel(const float* __restrict__ kb,
                 const float* __restrict__ vb,
                 const float* __restrict__ qb,
                 float* __restrict__ out) {
    const int lane = threadIdx.x & 31;
    const int wave = threadIdx.x >> 5;
    const int b    = blockIdx.y;
    const int i    = blockIdx.x * 8 + wave;     // state row index

    float s[16];
    #pragma unroll
    for (int c = 0; c < 16; ++c) s[c] = 0.f;

    for (int t = 0; t < TT; ++t) {
        const long base = ((long)t * BB + b) * NN;
        float4 kk[4], qq[4];
        #pragma unroll
        for (int c4 = 0; c4 < 4; ++c4) {
            const long off = base + c4 * 128 + lane * 4;
            kk[c4] = *(const float4*)(kb + off);
            qq[c4] = *(const float4*)(qb + off);
        }
        const float vi = vb[base + i];

        // prefetch next step's k/q rows into near caches (2KB each; one
        // byte per lane at 64B stride covers the row -> global_prefetch)
        {
            const int tn = (t + 1 < TT) ? (t + 1) : (TT - 1);
            const long basen = ((long)tn * BB + b) * NN + lane * 16;
            __builtin_prefetch(kb + basen, 0, 0);
            __builtin_prefetch(qb + basen, 0, 0);
        }

        // retrieved_i = S_i . k_t
        float r = 0.f;
        #pragma unroll
        for (int c4 = 0; c4 < 4; ++c4) {
            r = fmaf(s[c4*4+0], kk[c4].x, r);
            r = fmaf(s[c4*4+1], kk[c4].y, r);
            r = fmaf(s[c4*4+2], kk[c4].z, r);
            r = fmaf(s[c4*4+3], kk[c4].w, r);
        }
        r = wave_reduce_sum(r);
        const float delta = vi - r;

        // S_i <- tanh(S_i + delta * k_t)
        #pragma unroll
        for (int c4 = 0; c4 < 4; ++c4) {
            s[c4*4+0] = fast_tanh(fmaf(delta, kk[c4].x, s[c4*4+0]));
            s[c4*4+1] = fast_tanh(fmaf(delta, kk[c4].y, s[c4*4+1]));
            s[c4*4+2] = fast_tanh(fmaf(delta, kk[c4].z, s[c4*4+2]));
            s[c4*4+3] = fast_tanh(fmaf(delta, kk[c4].w, s[c4*4+3]));
        }
        // Sq_i = S_i . q_t
        float a = 0.f;
        #pragma unroll
        for (int c4 = 0; c4 < 4; ++c4) {
            a = fmaf(s[c4*4+0], qq[c4].x, a);
            a = fmaf(s[c4*4+1], qq[c4].y, a);
            a = fmaf(s[c4*4+2], qq[c4].z, a);
            a = fmaf(s[c4*4+3], qq[c4].w, a);
        }
        a = wave_reduce_sum(a);

        if (lane == 0) {
            // out = Sq * silu(Sq) = Sq^2 * sigmoid(Sq)
            const float sig = 1.0f / (1.0f + __expf(-a));
            out[base + i] = a * a * sig;
        }
    }

    // S_final, stored after the (T,B,N) outputs
    float* Sf = out + TBN + ((long)b * NN + i) * NN;
    #pragma unroll
    for (int c4 = 0; c4 < 4; ++c4) {
        float4 v4 = make_float4(s[c4*4+0], s[c4*4+1], s[c4*4+2], s[c4*4+3]);
        *(float4*)(Sf + c4 * 128 + lane * 4) = v4;
    }
}

// -------- launcher -------------------------------------------------------
extern "C" void kernel_launch(void* const* d_in, const int* in_sizes, int n_in,
                              void* d_out, int out_size, void* d_ws, size_t ws_size,
                              hipStream_t stream) {
    const float* x  = (const float*)d_in[0];
    const float* Wk = (const float*)d_in[1];
    const float* Wv = (const float*)d_in[2];
    const float* Wq = (const float*)d_in[3];
    float* out = (float*)d_out;

    // workspace: q | v | k  (each TBN fp32 = 16MB; total 48MB)
    float* qb = (float*)d_ws;
    float* vb = qb + TBN;
    float* kb = vb + TBN;

    // phase 1: three projections via split-bf16 WMMA
    dim3 pgrid(TB / 16, NN / 128, 3);
    proj_kernel<<<pgrid, 256, 0, stream>>>(x, Wq, Wv, Wk, qb, vb, kb);

    // phase 1b: normalize k rows in place
    knorm_kernel<<<TB / 8, 256, 0, stream>>>(kb);

    // phase 2: sequential scan, S resident in VGPRs
    dim3 sgrid(NN / 8, BB);
    scan_kernel<<<sgrid, 256, 0, stream>>>(kb, vb, qb, out);
}
